// CapsuleNet_42949672960932
// MI455X (gfx1250) — matrix operand order repaired and checked by reading, not compiled, not run on hardware
//
#include <hip/hip_runtime.h>
#include <hip/hip_bf16.h>
#include <math.h>

// ---------------------------------------------------------------------------
// CapsuleNet forward for MI455X (gfx1250, wave32, WMMA).
// Heavy convs are recast as GEMMs on v_wmma_f32_16x16x32_f16 (f16 operands,
// f32 accumulate). GEMM1 stages its A strip into LDS with the CDNA5 async
// DMA path (global_load_async_to_lds_b128 + s_wait_asynccnt) so each A row
// is fetched from L2 once per block instead of 16x. priors[10,32,8192,16]
// (168MB) is materialized in ws -- it fits the 192MB L2, so the 3 routing
// passes stay cache-resident; their reduction loops prefetch one strip ahead
// (global_prefetch_b8). Routing reductions use LDS; softmax/squash in f32.
// ---------------------------------------------------------------------------

typedef _Float16 half_t;
typedef __attribute__((ext_vector_type(16))) _Float16 v16h;
typedef __attribute__((ext_vector_type(8)))  float    v8f;

#define B_SZ 32
#define L_SZ 512
#define E_SZ 128
#define H_CH 256
#define NPOS 256          // L/2
#define NCAP 8192         // 32 * 256
#define NCLS 10
#define KW1  384          // 3*E

__constant__ int c_primes[8] = {3,5,7,9,11,13,17,19};
// prefix sums of p*32*256 (halfs) for packed primary weights
__constant__ int c_pbase[8]  = {0, 24576, 65536, 122880, 196608, 286720, 393216, 532480};

union FragH { v16h h; uint4 q[2]; };

// CDNA5 async copy: global -> LDS, 16B per lane, tracked by ASYNCcnt.
__device__ __forceinline__ void async_copy_b128(unsigned lds_off, const void* gaddr) {
    asm volatile("global_load_async_to_lds_b128 %0, %1, off"
                 :: "v"(lds_off), "v"(gaddr) : "memory");
}
__device__ __forceinline__ void wait_async0() {
    asm volatile("s_wait_asynccnt 0" ::: "memory");
}

// ---------------------------------------------------------------------------
// Build A1[B*L, 384] f16: row (b,t) = concat(emb[tok[t-1]], emb[tok[t]], emb[tok[t+1]])
// ---------------------------------------------------------------------------
__global__ void k_build_a1(const int* __restrict__ tok, const float* __restrict__ embed,
                           half_t* __restrict__ a1) {
    int idx = blockIdx.x * 256 + threadIdx.x;
    if (idx >= B_SZ * L_SZ * KW1) return;
    int j = idx % KW1, r = idx / KW1;
    int b = r / L_SZ, t = r % L_SZ;
    int seg = j / E_SZ, e = j % E_SZ;
    int tt = t + seg - 1;
    float v = 0.f;
    if (tt >= 0 && tt < L_SZ) {
        int token = tok[b * L_SZ + tt];
        v = embed[(size_t)token * E_SZ + e];
    }
    a1[idx] = (half_t)v;
}

// conv_w [256][384] f32 -> f16 (already [n][k] layout for B-fragments)
__global__ void k_pack_w1(const float* __restrict__ w, half_t* __restrict__ wh) {
    int idx = blockIdx.x * 256 + threadIdx.x;
    if (idx < H_CH * KW1) wh[idx] = (half_t)w[idx];
}

// w_prim_i [32][256][p] f32 -> wpk [k][c][cin] f16 (contiguous K per (k,c))
__global__ void k_pack_wp(const float* __restrict__ src, half_t* __restrict__ dst, int p) {
    int idx = blockIdx.x * 256 + threadIdx.x;
    if (idx >= p * 32 * 256) return;
    int k  = idx / (32 * 256);
    int rr = idx % (32 * 256);
    int c  = rr / 256;
    int ci = rr % 256;
    dst[idx] = (half_t)src[((size_t)c * 256 + ci) * p + k];
}

// pack 8 bias vectors (32 each) into one buffer
__global__ void k_pack_bp(const float* b0, const float* b1, const float* b2, const float* b3,
                          const float* b4, const float* b5, const float* b6, const float* b7,
                          float* __restrict__ bpk) {
    int idx = threadIdx.x;
    if (idx >= 256) return;
    int i = idx >> 5, c = idx & 31;
    const float* p;
    switch (i) {
        case 0: p = b0; break; case 1: p = b1; break;
        case 2: p = b2; break; case 3: p = b3; break;
        case 4: p = b4; break; case 5: p = b5; break;
        case 6: p = b6; break; default: p = b7; break;
    }
    bpk[idx] = p[c];
}

// ---------------------------------------------------------------------------
// GEMM1: h = relu(A1[16384,384] x W1[384,256] + b).
// One block per 16-row M-strip; A strip (12KB) DMA'd to LDS once (async),
// 8 waves x 2 N-tiles cover all 256 columns. Stores h as f16 [b*512+t][256].
// ---------------------------------------------------------------------------
__global__ void __launch_bounds__(256) k_gemm1(const half_t* __restrict__ a1,
                                               const half_t* __restrict__ w1h,
                                               const float* __restrict__ conv_b,
                                               half_t* __restrict__ h16) {
    __shared__ half_t shA[16 * KW1];   // 12 KB, row-major [16][384]
    int tid  = threadIdx.x;
    int row0 = blockIdx.x * 16;

    // Async DMA the contiguous 16x384 f16 strip into LDS (768 x 16B chunks).
    {
        const char* src = (const char*)(a1 + (size_t)row0 * KW1);
        unsigned    dst = (unsigned)(uintptr_t)shA;   // low 32 bits = LDS offset
        for (int j = 0; j < 3; ++j) {
            int chunk = tid + j * 256;
            async_copy_b128(dst + chunk * 16, src + chunk * 16);
        }
        wait_async0();          // drain this wave's ASYNCcnt
    }
    __syncthreads();            // all waves' DMA complete -> LDS valid

    int lane = tid & 31;
    int w    = tid >> 5;        // wave in block: 0..7
    int col0 = w * 32;
    int hi = lane >> 4;         // half-wave select
    int lm = lane & 15;

    v8f acc0 = {}, acc1 = {};
    const uint4*  aptr  = (const uint4*)(shA + (size_t)lm * KW1 + hi * 8);  // A: M=lm
    const half_t* bcol0 = w1h + (size_t)(col0 + lm) * KW1 + hi * 8;         // B: N=lm
    const half_t* bcol1 = w1h + (size_t)(col0 + 16 + lm) * KW1 + hi * 8;
    FragH A, B0, B1;
    for (int kk = 0; kk < KW1; kk += 32) {
        A.q[0]  = aptr[(kk >> 3)];       // ds_load_b128
        A.q[1]  = aptr[(kk >> 3) + 2];   // +16 halfs
        B0.q[0] = *(const uint4*)(bcol0 + kk);
        B0.q[1] = *(const uint4*)(bcol0 + kk + 16);
        B1.q[0] = *(const uint4*)(bcol1 + kk);
        B1.q[1] = *(const uint4*)(bcol1 + kk + 16);
        acc0 = __builtin_amdgcn_wmma_f32_16x16x32_f16(false, A.h, false, B0.h,
                                                      (short)0, acc0, false, false);
        acc1 = __builtin_amdgcn_wmma_f32_16x16x32_f16(false, A.h, false, B1.h,
                                                      (short)0, acc1, false, false);
    }
    int nn0 = col0 + lm, nn1 = col0 + 16 + lm;    // C/D: N = lane&15
    float bias0 = conv_b[nn0], bias1 = conv_b[nn1];
    for (int r = 0; r < 8; ++r) {
        int m = r + hi * 8;
        h16[(size_t)(row0 + m) * H_CH + nn0] = (half_t)fmaxf(acc0[r] + bias0, 0.f);
        h16[(size_t)(row0 + m) * H_CH + nn1] = (half_t)fmaxf(acc1[r] + bias1, 0.f);
    }
}

// ---------------------------------------------------------------------------
// Primary capsules: 8 convs (prime K, stride 2) as k-accumulated GEMMs.
// One wave handles (prime i, batch b, 16 positions, all 32 channels).
// Writes raw u[b][c*256+pos][i] f32.
// ---------------------------------------------------------------------------
__global__ void __launch_bounds__(256) k_caps(const half_t* __restrict__ h16,
                                              const half_t* __restrict__ wpk,
                                              const float* __restrict__ bpk,
                                              float* __restrict__ u) {
    int wid  = (blockIdx.x * 256 + threadIdx.x) >> 5;
    int lane = threadIdx.x & 31;
    int i    = wid >> 9;                       // 512 waves per prime
    int rem  = wid & 511;
    int b    = rem >> 4;
    int pos0 = (rem & 15) << 4;
    int p   = c_primes[i];
    int off = (p - 1) >> 1;
    const half_t* wb = wpk + c_pbase[i];
    int hi = lane >> 4;
    int lm = lane & 15;

    v8f acc0 = {}, acc1 = {};
    FragH A, B0, B1;
    int pos = pos0 + lm;                       // A row = position (stride 2 in t)
    for (int k = 0; k < p; ++k) {
        int t = 2 * pos + k - off;
        bool inb = (t >= 0) && (t < L_SZ);
        const half_t* arow = h16 + (size_t)(b * L_SZ + (inb ? t : 0)) * H_CH + hi * 8;
        for (int cc = 0; cc < H_CH; cc += 32) {
            if (inb) {
                A.q[0] = *(const uint4*)(arow + cc);
                A.q[1] = *(const uint4*)(arow + cc + 16);
            } else {
                A.q[0] = make_uint4(0, 0, 0, 0);
                A.q[1] = make_uint4(0, 0, 0, 0);
            }
            const half_t* wk = wb + ((size_t)k * 32) * H_CH + cc + hi * 8;
            B0.q[0] = *(const uint4*)(wk + (size_t)lm * H_CH);
            B0.q[1] = *(const uint4*)(wk + (size_t)lm * H_CH + 16);
            B1.q[0] = *(const uint4*)(wk + (size_t)(lm + 16) * H_CH);
            B1.q[1] = *(const uint4*)(wk + (size_t)(lm + 16) * H_CH + 16);
            acc0 = __builtin_amdgcn_wmma_f32_16x16x32_f16(false, A.h, false, B0.h,
                                                          (short)0, acc0, false, false);
            acc1 = __builtin_amdgcn_wmma_f32_16x16x32_f16(false, A.h, false, B1.h,
                                                          (short)0, acc1, false, false);
        }
    }
    float bias0 = bpk[i * 32 + lm];
    float bias1 = bpk[i * 32 + lm + 16];
    for (int r = 0; r < 8; ++r) {
        int m  = r + hi * 8;
        int po = pos0 + m;
        u[((size_t)b * NCAP + (size_t)lm * NPOS + po) * 8 + i]        = acc0[r] + bias0;
        u[((size_t)b * NCAP + (size_t)(lm + 16) * NPOS + po) * 8 + i] = acc1[r] + bias1;
    }
}

// squash over the 8-vector, in place
__global__ void k_squash(float* __restrict__ u) {
    int idx = blockIdx.x * 256 + threadIdx.x;
    if (idx >= B_SZ * NCAP) return;
    float* pu = u + (size_t)idx * 8;
    float t[8], s = 0.f;
    for (int i = 0; i < 8; ++i) { t[i] = pu[i]; s += t[i] * t[i]; }
    float sc = (s / (1.f + s)) * rsqrtf(fmaxf(s, 1e-30f));
    for (int i = 0; i < 8; ++i) pu[i] = t[i] * sc;
}

// priors[c][b][n][o] = sum_i u[b][n][i] * rw[c][n][i][o]   (K=8: VALU)
__global__ void k_priors(const float* __restrict__ u, const float* __restrict__ rw,
                         float* __restrict__ priors) {
    int idx = blockIdx.x * 256 + threadIdx.x;  // (c*32+b)*8192 + n
    if (idx >= NCLS * B_SZ * NCAP) return;
    int n  = idx & (NCAP - 1);
    int cb = idx >> 13;
    int b  = cb & 31, c = cb >> 5;
    float ui[8];
    const float* pu = u + ((size_t)b * NCAP + n) * 8;
    for (int i = 0; i < 8; ++i) ui[i] = pu[i];
    const float* w = rw + ((size_t)c * NCAP + n) * 128;
    float* po = priors + (size_t)idx * 16;
    for (int o = 0; o < 16; ++o) {
        float s = 0.f;
        for (int i = 0; i < 8; ++i) s += ui[i] * w[i * 16 + o];
        po[o] = s;
    }
}

// one routing iteration; one workgroup per (c,b)
__global__ void __launch_bounds__(256) k_route(const float* __restrict__ priors,
                                               float* __restrict__ logits,
                                               float* __restrict__ vbuf, int iter) {
    __shared__ float sh_p[NCAP];
    __shared__ float red[256];
    __shared__ float sv[16];
    int cb  = blockIdx.x;
    int tid = threadIdx.x;
    const float* pr = priors + (size_t)cb * NCAP * 16;
    float* lg = logits + (size_t)cb * NCAP;

    float invZ = 1.0f / (float)NCAP;
    if (iter > 0) {
        float mx = -1e30f;
        for (int n = tid; n < NCAP; n += 256) mx = fmaxf(mx, lg[n]);
        red[tid] = mx; __syncthreads();
        for (int s = 128; s > 0; s >>= 1) {
            if (tid < s) red[tid] = fmaxf(red[tid], red[tid + s]);
            __syncthreads();
        }
        float Mx = red[0]; __syncthreads();
        float sum = 0.f;
        for (int n = tid; n < NCAP; n += 256) {
            float e = __expf(lg[n] - Mx);
            sh_p[n] = e; sum += e;
        }
        red[tid] = sum; __syncthreads();
        for (int s = 128; s > 0; s >>= 1) {
            if (tid < s) red[tid] += red[tid + s];
            __syncthreads();
        }
        invZ = 1.0f / red[0]; __syncthreads();
    }

    float acc[16];
    for (int o = 0; o < 16; ++o) acc[o] = 0.f;
    for (int n = tid; n < NCAP; n += 256) {
        const float* pp = pr + (size_t)n * 16;
        __builtin_prefetch(pp + 256 * 16, 0, 3);   // global_prefetch_b8, next strip
        float pn = (iter == 0) ? invZ : sh_p[n] * invZ;
        for (int o = 0; o < 16; ++o) acc[o] += pn * pp[o];
    }
    for (int o = 0; o < 16; ++o) {
        red[tid] = acc[o]; __syncthreads();
        for (int s = 128; s > 0; s >>= 1) {
            if (tid < s) red[tid] += red[tid + s];
            __syncthreads();
        }
        if (tid == 0) sv[o] = red[0];
        __syncthreads();
    }
    if (tid == 0) {
        float sn = 0.f;
        for (int o = 0; o < 16; ++o) sn += sv[o] * sv[o];
        float sc = (sn / (1.f + sn)) * rsqrtf(fmaxf(sn, 1e-30f));
        for (int o = 0; o < 16; ++o) sv[o] *= sc;
    }
    __syncthreads();

    if (iter < 2) {
        for (int n = tid; n < NCAP; n += 256) {
            const float* pp = pr + (size_t)n * 16;
            __builtin_prefetch(pp + 256 * 16, 0, 3);
            float d = 0.f;
            for (int o = 0; o < 16; ++o) d += pp[o] * sv[o];
            lg[n] = (iter == 0) ? d : (lg[n] + d);   // iter0 writes (logits start at 0)
        }
    } else if (tid < 16) {
        vbuf[(size_t)cb * 16 + tid] = sv[tid];
    }
}

// class norms + softmax over C
__global__ void k_final(const float* __restrict__ vbuf, float* __restrict__ out) {
    int b = threadIdx.x;
    if (b >= B_SZ) return;
    float cls[NCLS];
    float mx = -1e30f;
    for (int c = 0; c < NCLS; ++c) {
        const float* v = vbuf + (size_t)(c * B_SZ + b) * 16;
        float s = 0.f;
        for (int o = 0; o < 16; ++o) s += v[o] * v[o];
        cls[c] = sqrtf(s);
        mx = fmaxf(mx, cls[c]);
    }
    float Z = 0.f;
    for (int c = 0; c < NCLS; ++c) { cls[c] = __expf(cls[c] - mx); Z += cls[c]; }
    float iZ = 1.0f / Z;
    for (int c = 0; c < NCLS; ++c) out[b * NCLS + c] = cls[c] * iZ;
}

// ---------------------------------------------------------------------------
extern "C" void kernel_launch(void* const* d_in, const int* in_sizes, int n_in,
                              void* d_out, int out_size, void* d_ws, size_t ws_size,
                              hipStream_t stream) {
    (void)in_sizes; (void)n_in; (void)out_size; (void)ws_size;
    const int*   tokens = (const int*)d_in[0];
    const float* embed  = (const float*)d_in[1];
    const float* conv_w = (const float*)d_in[2];
    const float* conv_b = (const float*)d_in[3];
    const float* wp[8];
    const float* bp[8];
    for (int i = 0; i < 8; ++i) {
        wp[i] = (const float*)d_in[4 + 2 * i];
        bp[i] = (const float*)d_in[5 + 2 * i];
    }
    const float* rw = (const float*)d_in[20];

    char* ws = (char*)d_ws;
    size_t off = 0;
    auto alloc = [&](size_t bytes) -> void* {
        off = (off + 255) & ~(size_t)255;
        void* p = ws + off;
        off += bytes;
        return p;
    };
    half_t* A1  = (half_t*)alloc((size_t)B_SZ * L_SZ * KW1 * 2);      // 12.6 MB
    half_t* W1H = (half_t*)alloc((size_t)H_CH * KW1 * 2);             // 0.2 MB
    half_t* WPK = (half_t*)alloc((size_t)688128 * 2);                 // 1.4 MB
    float*  BPK = (float*)alloc(256 * 4);
    half_t* H16 = (half_t*)alloc((size_t)B_SZ * L_SZ * H_CH * 2);     // 8.4 MB
    float*  U   = (float*)alloc((size_t)B_SZ * NCAP * 8 * 4);         // 8.4 MB
    float*  PR  = (float*)alloc((size_t)NCLS * B_SZ * NCAP * 16 * 4); // 168 MB (L2-resident)
    float*  LG  = (float*)alloc((size_t)NCLS * B_SZ * NCAP * 4);      // 10.5 MB
    float*  VB  = (float*)alloc((size_t)NCLS * B_SZ * 16 * 4);

    const int primes[8] = {3, 5, 7, 9, 11, 13, 17, 19};
    const int pbase[8]  = {0, 24576, 65536, 122880, 196608, 286720, 393216, 532480};

    k_build_a1<<<(B_SZ * L_SZ * KW1 + 255) / 256, 256, 0, stream>>>(tokens, embed, A1);
    k_pack_w1<<<(H_CH * KW1 + 255) / 256, 256, 0, stream>>>(conv_w, W1H);
    for (int i = 0; i < 8; ++i) {
        int n = primes[i] * 32 * 256;
        k_pack_wp<<<(n + 255) / 256, 256, 0, stream>>>(wp[i], WPK + pbase[i], primes[i]);
    }
    k_pack_bp<<<1, 256, 0, stream>>>(bp[0], bp[1], bp[2], bp[3], bp[4], bp[5], bp[6], bp[7], BPK);

    // one block per 16-row M-strip (A staged in LDS via async DMA)
    k_gemm1<<<1024, 256, 0, stream>>>(A1, W1H, conv_b, H16);
    // 8 primes * 32 batch * 16 pos-tiles = 4096 waves
    k_caps<<<512, 256, 0, stream>>>(H16, WPK, BPK, U);
    k_squash<<<(B_SZ * NCAP + 255) / 256, 256, 0, stream>>>(U);
    k_priors<<<(NCLS * B_SZ * NCAP + 255) / 256, 256, 0, stream>>>(U, rw, PR);
    for (int it = 0; it < 3; ++it)
        k_route<<<NCLS * B_SZ, 256, 0, stream>>>(PR, LG, VB, it);
    k_final<<<1, 32, 0, stream>>>(VB, (float*)d_out);
}
